// Net_59545426592369
// MI455X (gfx1250) — matrix operand order repaired and checked by reading, not compiled, not run on hardware
//
#include <hip/hip_runtime.h>
#include <hip/hip_bf16.h>

typedef __attribute__((ext_vector_type(16))) _Float16 v16h;
typedef __attribute__((ext_vector_type(8)))  float    v8f;

#define FEA 67
#define HID 16
#define OUTC 2
#define NSTK 3

// ---------------- elementwise helpers ----------------
__global__ void k_zero4(float4* __restrict__ p, int n4) {
    int i = blockIdx.x * blockDim.x + threadIdx.x;
    if (i < n4) p[i] = make_float4(0.f, 0.f, 0.f, 0.f);
}

__global__ void k_degree(const int* __restrict__ dst, float* __restrict__ deg, int E) {
    int e = blockIdx.x * blockDim.x + threadIdx.x;
    if (e < E) atomicAdd(&deg[dst[e]], 1.0f);
}

__global__ void k_dinv(float* __restrict__ deg, int N) {
    int i = blockIdx.x * blockDim.x + threadIdx.x;
    if (i < N) {
        float d = deg[i];
        deg[i] = (d > 0.0f) ? rsqrtf(fmaxf(d, 1.0f)) : 0.0f;
    }
}

__global__ void k_norm(const int* __restrict__ src, const int* __restrict__ dst,
                       const float* __restrict__ dinv, float* __restrict__ nm, int E) {
    int e = blockIdx.x * blockDim.x + threadIdx.x;
    if (e < E) nm[e] = dinv[src[e]] * dinv[dst[e]];
}

// ---------------- pack layer-1 weights into WMMA B-fragment layout (f16) ----------------
// Logical W: [96 (K, padded from 67)] x [96 (cols: 0..47 = init k*16+h, 48..95 = root)]
// Fragment layout: idx = ((s*6 + j)*32 + lane)*16 + h
//   K = s*32 + (lane>=16 ? 16 : 0) + h ;  Ncol = j*16 + (lane&15)
__global__ void k_pack_w1(const float* __restrict__ iw, const float* __restrict__ rw,
                          _Float16* __restrict__ Wf) {
    int idx = blockIdx.x * blockDim.x + threadIdx.x;
    if (idx >= 96 * 96) return;
    int h    = idx & 15;
    int lane = (idx >> 4) & 31;
    int sj   = idx >> 9;           // 0..17
    int s = sj / 6, j = sj % 6;
    int Kk   = s * 32 + ((lane >= 16) ? 16 : 0) + h;
    int Ncol = j * 16 + (lane & 15);
    float v = 0.0f;
    if (Kk < FEA) {
        if (Ncol < 48) { int k = Ncol >> 4, hh = Ncol & 15; v = iw[(k * FEA + Kk) * HID + hh]; }
        else { int c = Ncol - 48; int k = c >> 4, hh = c & 15; v = rw[(k * FEA + Kk) * HID + hh]; }
    }
    Wf[idx] = (_Float16)v;
}

// ---------------- WMMA GEMM: [N,67]f32 -> f16 -> @[67,96] -> O1[N][48], R1[N][48] ----------------
__global__ void __launch_bounds__(128)
k_gemm1_wmma(const float* __restrict__ x, const _Float16* __restrict__ Wf,
             float* __restrict__ O, float* __restrict__ R, int N, int mTiles) {
    __shared__ __align__(16) _Float16 sW[96 * 96];        // fragment layout
    __shared__ __align__(16) _Float16 sX[4][16 * 96];     // row-major x tile per wave
    const int tid  = threadIdx.x;
    const int wave = tid >> 5;
    const int lane = tid & 31;

    // stage weights (block-wide, vectorized 16B)
    {
        const uint4* wsrc = (const uint4*)Wf;
        uint4* wdst = (uint4*)sW;
        for (int i = tid; i < (96 * 96 * 2) / 16; i += 128) wdst[i] = wsrc[i];
    }

    const int mtile = blockIdx.x * 4 + wave;
    if (mtile < mTiles) {
        const int base = mtile * 16;
        for (int i = lane; i < 16 * 96; i += 32) {
            int r = i / 96, c = i - r * 96;
            int node = base + r;
            float v = (c < FEA && node < N) ? x[node * FEA + c] : 0.0f;
            sX[wave][i] = (_Float16)v;
        }
    }
    __syncthreads();
    if (mtile >= mTiles) return;

    v8f acc[6];
    const v8f zed = {0.f, 0.f, 0.f, 0.f, 0.f, 0.f, 0.f, 0.f};
#pragma unroll
    for (int j = 0; j < 6; ++j) acc[j] = zed;

    const int row = lane & 15;
    const int aHi = (lane >= 16) ? 16 : 0;  // A-matrix K sub-offset for upper half-wave

#pragma unroll
    for (int s = 0; s < 3; ++s) {
        const int Ks = s * 32;
        v16h afrag;
#pragma unroll
        for (int h = 0; h < 16; ++h) {
            int k = Ks + aHi + h;   // contiguous K order, matches packed B
            afrag[h] = sX[wave][row * 96 + k];
        }
#pragma unroll
        for (int j = 0; j < 6; ++j) {
            const v16h bfrag = *(const v16h*)&sW[(((s * 6 + j) * 32) + lane) * 16];
            acc[j] = __builtin_amdgcn_wmma_f32_16x16x32_f16(
                false, afrag, false, bfrag, (short)0, acc[j], false, false);
        }
    }

    // D layout: VGPR r -> M = r (+8 for lanes>=16), N = lane&15
    const int mOff = (lane < 16) ? 0 : 8;
    const int col0 = lane & 15;
    const int nodeBase = mtile * 16;
#pragma unroll
    for (int j = 0; j < 6; ++j) {
        const int col = j * 16 + col0;
#pragma unroll
        for (int r8 = 0; r8 < 8; ++r8) {
            const int node = nodeBase + mOff + r8;
            if (node < N) {
                const float v = acc[j][r8];
                if (col < 48) O[node * 48 + col] = v;
                else          R[node * 48 + (col - 48)] = v;
            }
        }
    }
}

// ---------------- edge propagation (C=48): one thread per (edge, float4 chunk) ----------------
__global__ void k_propagate48(const float* __restrict__ feat, const int* __restrict__ src,
                              const int* __restrict__ dst, const float* __restrict__ nm,
                              float* __restrict__ agg, int E) {
    long long i = (long long)blockIdx.x * blockDim.x + threadIdx.x;
    long long tot = (long long)E * 12;
    if (i >= tot) return;
    int e = (int)(i / 12);
    int q = (int)(i - (long long)e * 12);           // 0..11
    const float  w  = nm[e];
    const long long so = (long long)src[e] * 48 + q * 4;
    const long long doo = (long long)dst[e] * 48 + q * 4;
    const float4 f = *(const float4*)(feat + so);   // global_load_b128
    atomicAdd(&agg[doo + 0], w * f.x);
    atomicAdd(&agg[doo + 1], w * f.y);
    atomicAdd(&agg[doo + 2], w * f.z);
    atomicAdd(&agg[doo + 3], w * f.w);
}

// ---------------- edge propagation (C=6): one thread per (edge, float2 pair) ----------------
__global__ void k_propagate6(const float* __restrict__ feat, const int* __restrict__ src,
                             const int* __restrict__ dst, const float* __restrict__ nm,
                             float* __restrict__ agg, int E) {
    long long i = (long long)blockIdx.x * blockDim.x + threadIdx.x;
    long long tot = (long long)E * 3;
    if (i >= tot) return;
    int e = (int)(i / 3);
    int q = (int)(i - (long long)e * 3);            // 0..2
    const float  w  = nm[e];
    const long long so = (long long)src[e] * 6 + q * 2;
    const long long doo = (long long)dst[e] * 6 + q * 2;
    const float2 f = *(const float2*)(feat + so);   // 24B rows are float2-aligned
    atomicAdd(&agg[doo + 0], w * f.x);
    atomicAdd(&agg[doo + 1], w * f.y);
}

// ---------------- layer-1: fused relu(agg+root+b) @ w1 -> T ----------------
__global__ void k_combtrans1(const float* __restrict__ agg, const float* __restrict__ root,
                             const float* __restrict__ b1, const float* __restrict__ w1,
                             float* __restrict__ T, int N) {
    int i = blockIdx.x * blockDim.x + threadIdx.x;
    if (i >= N * 48) return;
    int n = i / 48, c = i - n * 48;
    int k = c >> 4, p = c & 15;
    const float* a = agg  + n * 48 + k * 16;
    const float* r = root + n * 48 + k * 16;
    const float* b = b1 + k * 16;
    const float* w = w1 + k * 256 + p;              // w1[k][h][p]
    float acc = 0.0f;
#pragma unroll
    for (int h = 0; h < HID; ++h) {
        float o = fmaxf(a[h] + r[h] + b[h], 0.0f);  // t=0 output (relu)
        acc = fmaf(o, w[h * 16], acc);
    }
    T[i] = acc;
}

__global__ void k_combine1_mean(const float* __restrict__ agg, const float* __restrict__ root,
                                const float* __restrict__ b1, float* __restrict__ hout, int N) {
    int i = blockIdx.x * blockDim.x + threadIdx.x;
    if (i >= N * 16) return;
    int n = i / 16, h = i - n * 16;
    float s = 0.0f;
#pragma unroll
    for (int k = 0; k < NSTK; ++k) {
        int c = k * 16 + h;
        s += fmaxf(agg[n * 48 + c] + root[n * 48 + c] + b1[c], 0.0f);
    }
    hout[i] = fmaxf(s * (1.0f / 3.0f), 0.0f);       // mean over stacks, outer relu
}

// ---------------- layer-2 dense (tiny) ----------------
__global__ void k_gemm2(const float* __restrict__ h16, const float* __restrict__ iw2,
                        const float* __restrict__ rw2, float* __restrict__ O2,
                        float* __restrict__ R2, int N) {
    int i = blockIdx.x * blockDim.x + threadIdx.x;
    if (i >= N * 12) return;
    int n = i / 12, c = i - n * 12;
    const float* hr = h16 + n * 16;
    bool isRoot = (c >= 6);
    int cc = isRoot ? c - 6 : c;
    int k = cc >> 1, o = cc & 1;
    const float* w = (isRoot ? rw2 : iw2) + (k * HID) * OUTC + o;   // w[k][h][o]
    float acc = 0.0f;
#pragma unroll
    for (int hh = 0; hh < HID; ++hh) acc = fmaf(hr[hh], w[hh * OUTC], acc);
    if (isRoot) R2[n * 6 + cc] = acc;
    else        O2[n * 6 + cc] = acc;
}

// ---------------- layer-2: fused (agg+root+b2) @ w2 -> T2 ----------------
__global__ void k_combtrans2(const float* __restrict__ agg, const float* __restrict__ root,
                             const float* __restrict__ b2, const float* __restrict__ w2,
                             float* __restrict__ T2, int N) {
    int i = blockIdx.x * blockDim.x + threadIdx.x;
    if (i >= N * 6) return;
    int n = i / 6, c = i - n * 6;
    int k = c >> 1, p = c & 1;
    int base = n * 6 + k * 2;
    float o0 = agg[base + 0] + root[base + 0] + b2[k * 2 + 0];   // t=0 out (no act)
    float o1 = agg[base + 1] + root[base + 1] + b2[k * 2 + 1];
    const float* w = w2 + k * 4;                    // w2[k][o][p]
    T2[i] = fmaf(o0, w[0 * 2 + p], o1 * w[1 * 2 + p]);
}

__global__ void k_final(const float* __restrict__ agg, const float* __restrict__ root,
                        const float* __restrict__ b2, float* __restrict__ out, int N) {
    int n = blockIdx.x * blockDim.x + threadIdx.x;
    if (n >= N) return;
    float o0 = 0.0f, o1 = 0.0f;
#pragma unroll
    for (int k = 0; k < NSTK; ++k) {
        o0 += agg[n * 6 + k * 2 + 0] + root[n * 6 + k * 2 + 0] + b2[k * 2 + 0];
        o1 += agg[n * 6 + k * 2 + 1] + root[n * 6 + k * 2 + 1] + b2[k * 2 + 1];
    }
    o0 *= (1.0f / 3.0f);
    o1 *= (1.0f / 3.0f);
    float m = fmaxf(o0, o1);
    float lse = m + logf(expf(o0 - m) + expf(o1 - m));
    out[n * 2 + 0] = o0 - lse;
    out[n * 2 + 1] = o1 - lse;
}

// ---------------- launch ----------------
extern "C" void kernel_launch(void* const* d_in, const int* in_sizes, int n_in,
                              void* d_out, int out_size, void* d_ws, size_t ws_size,
                              hipStream_t stream) {
    const float* x    = (const float*)d_in[0];
    const int*   ei   = (const int*)d_in[1];
    const float* iw1  = (const float*)d_in[2];
    const float* w1   = (const float*)d_in[3];
    const float* rw1  = (const float*)d_in[4];
    const float* b1   = (const float*)d_in[5];
    const float* iw2  = (const float*)d_in[6];
    const float* w2   = (const float*)d_in[7];
    const float* rw2  = (const float*)d_in[8];
    const float* b2   = (const float*)d_in[9];
    float* out = (float*)d_out;

    const int N = in_sizes[0] / FEA;
    const int E = in_sizes[1] / 2;
    const int* src = ei;
    const int* dst = ei + E;

    // workspace layout (floats, all offsets multiple of 4 -> 16B aligned)
    float* ws = (float*)d_ws;
    size_t off = 0;
    float*     deg  = ws + off; off += (size_t)N;            // degree -> dinv (in place)
    float*     nrm  = ws + off; off += (size_t)E;
    _Float16*  Wf   = (_Float16*)(ws + off); off += (96 * 96) / 2;
    float*     O1   = ws + off; off += (size_t)N * 48;
    float*     R1   = ws + off; off += (size_t)N * 48;
    float*     AG1  = ws + off; off += (size_t)N * 48;
    float*     T1   = ws + off; off += (size_t)N * 48;
    float*     H16  = ws + off; off += (size_t)N * 16;
    float*     O2   = ws + off; off += (size_t)N * 6;
    float*     R2   = ws + off; off += (size_t)N * 6;
    float*     AG2  = ws + off; off += (size_t)N * 6;
    float*     T2   = ws + off; off += (size_t)N * 6;
    (void)ws_size; (void)n_in; (void)out_size;

    const int B = 256;
    auto g = [&](long long n) { return (unsigned)((n + B - 1) / B); };

    // --- gcn_norm ---
    k_zero4<<<g(N / 4), B, 0, stream>>>((float4*)deg, N / 4);
    k_degree<<<g(E), B, 0, stream>>>(dst, deg, E);
    k_dinv<<<g(N), B, 0, stream>>>(deg, N);
    k_norm<<<g(E), B, 0, stream>>>(src, dst, deg, nrm, E);

    // --- layer 1: fused WMMA GEMM for init + root ---
    k_pack_w1<<<g(96 * 96), B, 0, stream>>>(iw1, rw1, Wf);
    const int mTiles = (N + 15) / 16;
    k_gemm1_wmma<<<(mTiles + 3) / 4, 128, 0, stream>>>(x, Wf, O1, R1, N, mTiles);

    // t = 0: propagate out0, then fused relu-combine + stack transform -> T1
    k_zero4<<<g((long long)N * 12), B, 0, stream>>>((float4*)AG1, N * 12);
    k_propagate48<<<g((long long)E * 12), B, 0, stream>>>(O1, src, dst, nrm, AG1, E);
    k_combtrans1<<<g((long long)N * 48), B, 0, stream>>>(AG1, R1, b1, w1, T1, N);
    // t = 1: propagate T1, combine + mean over stacks -> H16
    k_zero4<<<g((long long)N * 12), B, 0, stream>>>((float4*)AG1, N * 12);
    k_propagate48<<<g((long long)E * 12), B, 0, stream>>>(T1, src, dst, nrm, AG1, E);
    k_combine1_mean<<<g((long long)N * 16), B, 0, stream>>>(AG1, R1, b1, H16, N);

    // --- layer 2 ---
    k_gemm2<<<g((long long)N * 12), B, 0, stream>>>(H16, iw2, rw2, O2, R2, N);
    // t = 0
    k_zero4<<<g((long long)N * 6 / 4), B, 0, stream>>>((float4*)AG2, (int)((long long)N * 6 / 4));
    k_propagate6<<<g((long long)E * 3), B, 0, stream>>>(O2, src, dst, nrm, AG2, E);
    k_combtrans2<<<g((long long)N * 6), B, 0, stream>>>(AG2, R2, b2, w2, T2, N);
    // t = 1
    k_zero4<<<g((long long)N * 6 / 4), B, 0, stream>>>((float4*)AG2, (int)((long long)N * 6 / 4));
    k_propagate6<<<g((long long)E * 3), B, 0, stream>>>(T2, src, dst, nrm, AG2, E);
    k_final<<<g(N), B, 0, stream>>>(AG2, R2, b2, out, N);
}
